// MultiheadAttention_4002909520710
// MI455X (gfx1250) — compile-verified, matrix operand order checked by reading, not run on hardware
//
#include <hip/hip_runtime.h>
#include <math.h>

typedef _Float16 half_t;
typedef __attribute__((ext_vector_type(16))) _Float16 v16h;
typedef __attribute__((ext_vector_type(8)))  _Float16 v8h;
typedef __attribute__((ext_vector_type(8)))  float    v8f;

#define S_LEN 2048
#define EMB   512
#define NH    8
#define HD    64
#define NB    8    // batch
#define KT    64   // kv tile

#define USE_ASYNC_LDS 1

// ---------------------------------------------------------------------------
// Fragment loaders.
// CDNA5 16-bit A/B fragment (16x32, wave32): lane L (r = L&15, hi = L>>4)
// holds row r with halves = K[hi*8 .. hi*8+7] and K[hi*8+16 .. hi*8+23].
// Caller passes the row pointer already offset by (frag*32 + hi*8) halves.
// ---------------------------------------------------------------------------
__device__ __forceinline__ v16h load_frag16(const half_t* p) {
  union { v16h v; v8h h[2]; } u;
  u.h[0] = *(const v8h*)(p);
  u.h[1] = *(const v8h*)(p + 16);
  return u.v;
}

__device__ __forceinline__ v16h load_frag_f32cvt(const float* p) {
  union { v16h v; half_t e[16]; } u;
#pragma unroll
  for (int i = 0; i < 8; ++i) {
    u.e[i]     = (half_t)p[i];
    u.e[i + 8] = (half_t)p[i + 16];
  }
  return u.v;
}

__device__ __forceinline__ v8f wmma_f16(v16h a, v16h b, v8f c) {
  // D = A(16x32 f16) * B(32x16 f16) + C(f32)
  return __builtin_amdgcn_wmma_f32_16x16x32_f16(false, a, false, b,
                                                (short)0, c, false, false);
}

__device__ __forceinline__ float hgroup_max(float v) {
#pragma unroll
  for (int m = 1; m <= 8; m <<= 1) v = fmaxf(v, __shfl_xor(v, m, 32));
  return v;  // reduced within each 16-lane half of the wave
}
__device__ __forceinline__ float hgroup_sum(float v) {
#pragma unroll
  for (int m = 1; m <= 8; m <<= 1) v += __shfl_xor(v, m, 32);
  return v;
}

// ---------------------------------------------------------------------------
// CDNA5 async global->LDS copy (ASYNCcnt path, CDNA5 ISA).
// lds_off: per-lane LDS byte offset (generic LDS addr truncates to offset).
// ---------------------------------------------------------------------------
__device__ __forceinline__ void async_copy_b128(unsigned lds_off,
                                                const half_t* gptr) {
#if USE_ASYNC_LDS
  asm volatile("global_load_async_to_lds_b128 %0, %1, off"
               :: "v"(lds_off), "v"(gptr)
               : "memory");
#else
  // fallback: through-VGPR copy
  *(v8h*)((char*)nullptr + lds_off) = *(const v8h*)gptr;  // unused
#endif
}
__device__ __forceinline__ void async_wait0() {
#if USE_ASYNC_LDS
  asm volatile("s_wait_asynccnt 0x0" ::: "memory");
#endif
}

// ---------------------------------------------------------------------------
// Kernel 1: f32 -> f16 weight conversion
// ---------------------------------------------------------------------------
__global__ void cvt_f32_f16(const float* __restrict__ src,
                            half_t* __restrict__ dst, int n) {
  int i = blockIdx.x * 256 + threadIdx.x;
  if (i < n) dst[i] = (half_t)src[i];
}

// ---------------------------------------------------------------------------
// Kernel 2: pack int32 mask -> 1 bit (4MB packed; L2-resident across heads)
// ---------------------------------------------------------------------------
__global__ void pack_mask(const int* __restrict__ mask,
                          unsigned* __restrict__ packed, int nwords) {
  int w = blockIdx.x * 256 + threadIdx.x;
  if (w >= nwords) return;
  const int* src = mask + (size_t)w * 32;
  unsigned bits = 0;
#pragma unroll
  for (int i = 0; i < 32; ++i) bits |= (src[i] ? 1u : 0u) << i;
  packed[w] = bits;
}

// ---------------------------------------------------------------------------
// Kernel 3: QKV projection.  out = X @ W^T + bias -> f16.
// One wave per 16-row x 64-col tile; all 4 weight fragments preloaded per
// K-step so the 8 B128 loads are in flight before the 4-WMMA burst.
// Q,K stored [b][h][s][64]; V stored transposed [b][h][64][s].
// ---------------------------------------------------------------------------
__global__ void qkv_proj(const float* __restrict__ xq,
                         const float* __restrict__ xkv,
                         const half_t* __restrict__ w16,   // [3][512][512]
                         const float* __restrict__ bq,
                         const float* __restrict__ bk,
                         const float* __restrict__ bv,
                         half_t* __restrict__ Qo,
                         half_t* __restrict__ Ko,
                         half_t* __restrict__ Vt) {
  const int lane = threadIdx.x & 31;
  const int hi = lane >> 4, lo = lane & 15;
  const int row0 = blockIdx.x * 16;      // over B*S = 16384 rows
  const int htile = blockIdx.y;          // head (column group / 64)
  const int which = blockIdx.z;          // 0=Q 1=K 2=V

  const float*  x    = (which == 0) ? xq : xkv;
  const half_t* w    = w16 + (size_t)which * EMB * EMB;
  const float*  bias = (which == 0) ? bq : ((which == 1) ? bk : bv);

  const float*  xrow  = x + (size_t)(row0 + lo) * EMB;
  const half_t* wbase = w + (size_t)(htile * HD + lo) * EMB + hi * 8;

  v8f acc[4] = {};
#pragma unroll 2
  for (int k0 = 0; k0 < EMB; k0 += 32) {
    v16h wf[4];
#pragma unroll
    for (int ns = 0; ns < 4; ++ns)
      wf[ns] = load_frag16(wbase + (size_t)(ns * 16) * EMB + k0);
    v16h afrag = load_frag_f32cvt(xrow + k0 + hi * 8);
#pragma unroll
    for (int ns = 0; ns < 4; ++ns)
      acc[ns] = wmma_f16(afrag, wf[ns], acc[ns]);
  }

  const int b = row0 >> 11;              // S_LEN = 2048
  const int s0 = row0 & (S_LEN - 1);
#pragma unroll
  for (int ns = 0; ns < 4; ++ns) {
    const int d = ns * 16 + lo;
    const float bv_ = bias[htile * HD + d];
#pragma unroll
    for (int g = 0; g < 8; ++g) {
      const int s = s0 + g + 8 * hi;     // D-layout: M = g + 8*hi, N = lo
      const half_t val = (half_t)(acc[ns][g] + bv_);
      if (which == 0)
        Qo[(((size_t)b * NH + htile) * S_LEN + s) * HD + d] = val;
      else if (which == 1)
        Ko[(((size_t)b * NH + htile) * S_LEN + s) * HD + d] = val;
      else
        Vt[(((size_t)b * NH + htile) * HD + d) * S_LEN + s] = val;
    }
  }
}

// ---------------------------------------------------------------------------
// Kernel 4: flash attention.  4 waves / block; block covers (b, h, 64 q rows).
// Each 64x64 K and V tile is staged ONCE into LDS via async global->LDS
// copies and shared by all 4 waves (4x cut in L2 read traffic), then consumed
// as ds_load_b128 fragments.  Online softmax in f32 registers.
// ---------------------------------------------------------------------------
__global__ void flash_attn(const half_t* __restrict__ Q,
                           const half_t* __restrict__ K,
                           const half_t* __restrict__ Vt,
                           const unsigned* __restrict__ pmask,
                           float* __restrict__ out) {
  __shared__ __align__(16) half_t Ks[KT * HD];       // [r][d]   8KB
  __shared__ __align__(16) half_t Vs[HD * KT];       // [d][r]   8KB
  __shared__ __align__(16) half_t Ps[4][16 * KT];    // per-wave 8KB

  const int tid  = threadIdx.x;
  const int wid  = tid >> 5;
  const int lane = tid & 31;
  const int hi = lane >> 4, lo = lane & 15;
  const int q0 = blockIdx.x * 64 + wid * 16;
  const int h  = blockIdx.y;
  const int b  = blockIdx.z;

  const half_t* Qb = Q  + (((size_t)b * NH + h) * S_LEN + q0) * HD;
  const half_t* Kb = K  +  ((size_t)b * NH + h) * S_LEN * HD;
  const half_t* Vb = Vt +  ((size_t)b * NH + h) * HD * S_LEN;
  const unsigned* mrow = pmask + ((size_t)b * S_LEN + q0) * (S_LEN / 32);

  const unsigned ks_base = (unsigned)(size_t)(void*)&Ks[0];
  const unsigned vs_base = (unsigned)(size_t)(void*)&Vs[0];

  // Q A-fragments (rows q0+lo), K-dim 64 as two 32-chunks
  v16h qf[2];
  qf[0] = load_frag16(Qb + lo * HD + 0  + hi * 8);
  qf[1] = load_frag16(Qb + lo * HD + 32 + hi * 8);

  v8f O[4] = {};
  float mrun[8], lrun[8];
#pragma unroll
  for (int g = 0; g < 8; ++g) { mrun[g] = -3.0e38f; lrun[g] = 0.0f; }

  for (int k0 = 0; k0 < S_LEN; k0 += KT) {
    // ---- stage K tile (contiguous 8KB) and V tile (64 row slices) ----
#pragma unroll
    for (int i = 0; i < 4; ++i) {
      const int c = tid + 128 * i;                 // 0..511 b128 chunks
      async_copy_b128(ks_base + c * 16, Kb + (size_t)k0 * HD + c * 8);
      const int r = c >> 3, cc = c & 7;
      async_copy_b128(vs_base + c * 16,
                      Vb + (size_t)r * S_LEN + k0 + cc * 8);
    }
    if (k0 + KT < S_LEN) {                         // prefetch next tiles
      if (tid < 64)
        __builtin_prefetch(Kb + (size_t)(k0 + KT) * HD + tid * 64, 0, 0);
      __builtin_prefetch(Vb + (size_t)(tid & 63) * S_LEN + k0 + KT, 0, 0);
    }
    async_wait0();
    __syncthreads();                               // tile visible to block

    // ---- scores S = (Q K^T)/8 from LDS fragments ----
    v8f sf[4];
#pragma unroll
    for (int ns = 0; ns < 4; ++ns) {
      sf[ns] = (v8f){};
#pragma unroll
      for (int kk = 0; kk < 2; ++kk) {
        const half_t* krow = &Ks[(ns * 16 + lo) * HD + kk * 32 + hi * 8];
        sf[ns] = wmma_f16(qf[kk], load_frag16(krow), sf[ns]);
      }
    }

    // ---- mask (2 packed words per row per tile) ----
#pragma unroll
    for (int g = 0; g < 8; ++g) {
      const unsigned* mw = &mrow[(g + 8 * hi) * (S_LEN / 32) + (k0 >> 5)];
      const unsigned wm0 = mw[0], wm1 = mw[1];
#pragma unroll
      for (int ns = 0; ns < 4; ++ns) {
        float s = sf[ns][g] * 0.125f;
        const unsigned wm = (ns & 2) ? wm1 : wm0;
        if ((wm >> ((ns & 1) * 16 + lo)) & 1u) s = -1.0e9f;
        sf[ns][g] = s;
      }
    }

    // ---- online softmax update ----
    float alpha[8];
#pragma unroll
    for (int g = 0; g < 8; ++g) {
      float t = fmaxf(fmaxf(sf[0][g], sf[1][g]), fmaxf(sf[2][g], sf[3][g]));
      t = hgroup_max(t);
      const float mn = fmaxf(mrun[g], t);
      alpha[g] = __expf(mrun[g] - mn);
      mrun[g] = mn;
    }
#pragma unroll
    for (int ns = 0; ns < 4; ++ns)
#pragma unroll
      for (int g = 0; g < 8; ++g)
        sf[ns][g] = __expf(sf[ns][g] - mrun[g]);
#pragma unroll
    for (int g = 0; g < 8; ++g) {
      float t = sf[0][g] + sf[1][g] + sf[2][g] + sf[3][g];
      t = hgroup_sum(t);
      lrun[g] = lrun[g] * alpha[g] + t;
    }

    // ---- P tile (D-layout) -> per-wave LDS -> A-layout fragments ----
    half_t* P = &Ps[wid][0];
#pragma unroll
    for (int ns = 0; ns < 4; ++ns)
#pragma unroll
      for (int g = 0; g < 8; ++g)
        P[(g + 8 * hi) * KT + ns * 16 + lo] = (half_t)sf[ns][g];
    asm volatile("" ::: "memory");   // keep DS store->load order (same wave)

#pragma unroll
    for (int ds = 0; ds < 4; ++ds)
#pragma unroll
      for (int g = 0; g < 8; ++g)
        O[ds][g] *= alpha[g];

    v16h pf[2];
    pf[0] = load_frag16(&P[lo * KT + 0  + hi * 8]);
    pf[1] = load_frag16(&P[lo * KT + 32 + hi * 8]);
#pragma unroll
    for (int ds = 0; ds < 4; ++ds) {
#pragma unroll
      for (int kk = 0; kk < 2; ++kk) {
        const half_t* vrow = &Vs[(ds * 16 + lo) * KT + kk * 32 + hi * 8];
        O[ds] = wmma_f16(pf[kk], load_frag16(vrow), O[ds]);
      }
    }
    __syncthreads();   // all waves done with Ks/Vs before next staging
  }

  // ---- epilogue: normalize and store f32 ----
  float invl[8];
#pragma unroll
  for (int g = 0; g < 8; ++g) invl[g] = 1.0f / lrun[g];
#pragma unroll
  for (int ds = 0; ds < 4; ++ds)
#pragma unroll
    for (int g = 0; g < 8; ++g) {
      const int s = q0 + g + 8 * hi;
      out[((size_t)b * S_LEN + s) * (NH * HD) + h * HD + ds * 16 + lo] =
          O[ds][g] * invl[g];
    }
}

// ---------------------------------------------------------------------------
extern "C" void kernel_launch(void* const* d_in, const int* in_sizes, int n_in,
                              void* d_out, int out_size, void* d_ws,
                              size_t ws_size, hipStream_t stream) {
  const float* x_q  = (const float*)d_in[0];
  const float* x_kv = (const float*)d_in[1];
  const int*   amsk = (const int*)  d_in[2];
  const float* w_q  = (const float*)d_in[3];
  const float* b_q  = (const float*)d_in[4];
  const float* w_k  = (const float*)d_in[5];
  const float* b_k  = (const float*)d_in[6];
  const float* w_v  = (const float*)d_in[7];
  const float* b_v  = (const float*)d_in[8];
  float* out = (float*)d_out;

  char* ws = (char*)d_ws;
  const size_t W_BYTES  = (size_t)3 * EMB * EMB * sizeof(half_t);        // 1.5MB
  const size_t QK_BYTES = (size_t)NB * NH * S_LEN * HD * sizeof(half_t); // 16MB
  half_t*   W16 = (half_t*)(ws);
  half_t*   Q16 = (half_t*)(ws + W_BYTES);
  half_t*   K16 = (half_t*)(ws + W_BYTES + QK_BYTES);
  half_t*   Vt  = (half_t*)(ws + W_BYTES + 2 * QK_BYTES);
  unsigned* PM  = (unsigned*)(ws + W_BYTES + 3 * QK_BYTES);              // 4MB

  // 1) weights -> f16
  const int wn = EMB * EMB;
  cvt_f32_f16<<<(wn + 255) / 256, 256, 0, stream>>>(w_q, W16 + 0 * wn, wn);
  cvt_f32_f16<<<(wn + 255) / 256, 256, 0, stream>>>(w_k, W16 + 1 * wn, wn);
  cvt_f32_f16<<<(wn + 255) / 256, 256, 0, stream>>>(w_v, W16 + 2 * wn, wn);

  // 2) mask -> bit pack
  const int nwords = NB * S_LEN * (S_LEN / 32);
  pack_mask<<<(nwords + 255) / 256, 256, 0, stream>>>(amsk, PM, nwords);

  // 3) QKV projection: 1024 row tiles x 8 heads x {Q,K,V}, 1 wave each
  qkv_proj<<<dim3((NB * S_LEN) / 16, NH, 3), 32, 0, stream>>>(
      x_q, x_kv, W16, b_q, b_k, b_v, Q16, K16, Vt);

  // 4) flash attention: 32 q-blocks x 8 heads x 8 batch, 4 waves each
  flash_attn<<<dim3(S_LEN / KT, NH, NB), 128, 0, stream>>>(
      Q16, K16, Vt, PM, out);
}
